// GNNModel_3848290697329
// MI455X (gfx1250) — matrix operand order repaired and checked by reading, not compiled, not run on hardware
//
#include <hip/hip_runtime.h>

// Problem constants (from reference)
#define HID 512
static const int ND = 124, NI = 177, NT = 104;

typedef __attribute__((ext_vector_type(16))) __bf16 v16bf;
typedef __attribute__((ext_vector_type(8)))  float  v8f;
typedef __attribute__((ext_vector_type(4)))  int    vi4;
typedef __attribute__((ext_vector_type(4)))  float  vf4;
typedef __attribute__((ext_vector_type(4)))  unsigned int uv4;
typedef __attribute__((ext_vector_type(8)))  int    iv8;

union Frag { v16bf v; vi4 q[2]; };

#if __has_builtin(__builtin_amdgcn_tensor_load_to_lds) && \
    __has_builtin(__builtin_amdgcn_s_wait_tensorcnt)
#define USE_TDM 1
#else
#define USE_TDM 0
#endif

// fp32 -> bf16 bits, round-to-nearest-even
__device__ __forceinline__ unsigned short f2bf(float f) {
  unsigned u = __float_as_uint(f);
  u += 0x7FFFu + ((u >> 16) & 1u);
  return (unsigned short)(u >> 16);
}

__device__ __forceinline__ unsigned pk2bf(float a, float b) {
  return (unsigned)f2bf(a) | ((unsigned)f2bf(b) << 16);
}

#if USE_TDM
// Issue a 2-D TDM load: tile (tile_x elems/row, tile_y rows) of fp32 starting at
// gp, row stride `stride` elems, into LDS byte offset lds_off. Elements beyond
// (tdim_x, tdim_y) read as zero (hardware zero-padding). D# layout per CDNA5
// ISA ch.8 (groups 2/3 NULL). 6-arg builtin form on this toolchain:
//   (uint32x4 g0, int32x8 g1, int32x4 g2, int32x4 g3, int32x8 g4, i32 cpol)
__device__ __forceinline__ void tdm_load_2d(const void* gp, unsigned lds_off,
                                            unsigned tile_x, unsigned tile_y,
                                            unsigned tdim_x, unsigned tdim_y,
                                            unsigned stride) {
  unsigned long long ga = (unsigned long long)gp;
  uv4 g0 = {0u, 0u, 0u, 0u};
  g0[0] = 1u;                                            // count=1 (valid D#)
  g0[1] = lds_off;                                       // lds_addr (bytes)
  g0[2] = (unsigned)ga;                                  // global_addr[31:0]
  g0[3] = (unsigned)((ga >> 32) & 0x01FFFFFFu) | (2u << 30);  // addr[56:32], type=2
  iv8 g1 = {0, 0, 0, 0, 0, 0, 0, 0};
  g1[0] = (int)(2u << 16);                               // data_size = 4B
  g1[1] = (int)((tdim_x & 0xFFFFu) << 16);               // tensor_dim0[15:0]
  g1[2] = (int)(((tdim_x >> 16) & 0xFFFFu) | ((tdim_y & 0xFFFFu) << 16));
  g1[3] = (int)(((tdim_y >> 16) & 0xFFFFu) | ((tile_x & 0xFFFFu) << 16));
  g1[4] = (int)(tile_y & 0xFFFFu);                       // tile_dim1 (tile_dim2=0)
  g1[5] = (int)stride;                                   // tensor_dim0_stride[31:0]
  vi4 z4 = {0, 0, 0, 0};
  iv8 z8 = {0, 0, 0, 0, 0, 0, 0, 0};
  __builtin_amdgcn_tensor_load_to_lds(g0, g1, z4, z4, z8, 0);
}
#endif

// ---------------------------------------------------------------------------
// C[M,N] = A[M,K] @ B[K,N] (+bias). Row-major fp32; N multiple of 128.
// Block: 256 threads (8 waves). Tile: 32(M) x 128(N), K stepped by 32.
// Both A and B tiles are TDM-prefetched (double buffered) into LDS as raw
// fp32 with hardware zero-padding at the M edge and K tail, then converted to
// bf16 fragment layout. Fragments load as 2x ds_load_b128 per operand.
// ---------------------------------------------------------------------------
__global__ __launch_bounds__(256) void wmma_gemm_kernel(
    const float* __restrict__ A, const float* __restrict__ B,
    const float* __restrict__ bias, float* __restrict__ C,
    int M, int K, int N) {
  __shared__ __align__(16) char smem[53760];
  float* bS0 = (float*)smem;                              // 32x128 fp32 (16KB)
  float* bS1 = (float*)(smem + 16384);                    // 32x128 fp32 (16KB)
  float* aS0 = (float*)(smem + 32768);                    // 32x32 fp32 (4KB)
  float* aS1 = (float*)(smem + 36864);                    // 32x32 fp32 (4KB)
  unsigned short* bH = (unsigned short*)(smem + 40960);   // 128 x 40 bf16
  unsigned short* aH = (unsigned short*)(smem + 51200);   // 32 x 40 bf16
  const int* aTi = (const int*)aH;
  const int* bTi = (const int*)bH;

  const int tid  = threadIdx.x;
  const int lane = tid & 31;
  const int wave = tid >> 5;
  const int m0   = blockIdx.x * 32;
  const int n0   = blockIdx.y * 128;
  const int KT   = (K + 31) >> 5;        // number of 32-wide K tiles
  const bool w0  = (__builtin_amdgcn_readfirstlane(wave) == 0);

  v8f c0 = {}, c1 = {};

#if USE_TDM
  if (w0) {
    tdm_load_2d(A + (size_t)m0 * K, 32768u, 32u, 32u,
                (unsigned)K, (unsigned)(M - m0), (unsigned)K);
    tdm_load_2d(B + n0, 0u, 128u, 32u, 128u, (unsigned)K, (unsigned)N);
  }
#endif

  int buf = 0;
  for (int t = 0; t < KT; ++t) {
    const int kt = t << 5;
#if USE_TDM
    if (w0) {
      if (t + 1 < KT) {
        tdm_load_2d(A + (size_t)m0 * K + kt + 32, buf ? 32768u : 36864u,
                    32u, 32u, (unsigned)(K - kt - 32), (unsigned)(M - m0),
                    (unsigned)K);
        tdm_load_2d(B + (size_t)(kt + 32) * N + n0, buf ? 0u : 16384u,
                    128u, 32u, 128u, (unsigned)(K - kt - 32), (unsigned)N);
        __builtin_amdgcn_s_wait_tensorcnt(2);  // current (A,B) pair complete
      } else {
        __builtin_amdgcn_s_wait_tensorcnt(0);
      }
    }
#else
    {
      float* aSd = buf ? aS1 : aS0;
      float* bSd = buf ? bS1 : bS0;
#pragma unroll
      for (int e = 0; e < 4; ++e) {
        int idx = e * 256 + tid;
        int m = idx >> 5, k = idx & 31;
        int gm = m0 + m, gk = kt + k;
        int gmc = gm < M ? gm : M - 1;
        int gkc = gk < K ? gk : K - 1;
        float v = A[(size_t)gmc * K + gkc];
        aSd[m * 32 + k] = (gm < M && gk < K) ? v : 0.0f;
      }
      vf4 vz = {0.0f, 0.0f, 0.0f, 0.0f};
#pragma unroll
      for (int e = 0; e < 4; ++e) {
        int vidx = e * 256 + tid;
        int k = vidx >> 5, n4 = (vidx & 31) << 2;
        int gk = kt + k;
        int gkc = gk < K ? gk : K - 1;
        vf4 v = *(const vf4*)&B[(size_t)gkc * N + n0 + n4];
        *(vf4*)&bSd[k * 128 + n4] = (gk < K) ? v : vz;
      }
    }
#endif
    __syncthreads();  // staged tiles ready; also protects bH/aH from prior reads

    const float* aSc = buf ? aS1 : aS0;
    const float* bSc = buf ? bS1 : bS0;
    // A convert: 32x32 fp32 -> bf16 aH[m*40 + k]; one vec4 + b64 store/thread
    {
      int m = tid >> 3, k4 = (tid & 7) << 2;
      vf4 v = *(const vf4*)&aSc[m * 32 + k4];
      uint2 p;
      p.x = pk2bf(v[0], v[1]);
      p.y = pk2bf(v[2], v[3]);
      *(uint2*)&aH[m * 40 + k4] = p;
    }
    // B convert: 32x128 fp32 -> bf16 fragment layout bH[n*40 + k]
#pragma unroll
    for (int e = 0; e < 4; ++e) {
      int vidx = e * 256 + tid;           // 1024 float4s
      int k = vidx >> 5, n4 = (vidx & 31) << 2;
      vf4 v = *(const vf4*)&bSc[k * 128 + n4];
      bH[(n4 + 0) * 40 + k] = f2bf(v[0]);
      bH[(n4 + 1) * 40 + k] = f2bf(v[1]);
      bH[(n4 + 2) * 40 + k] = f2bf(v[2]);
      bH[(n4 + 3) * 40 + k] = f2bf(v[3]);
    }
    __syncthreads();

    Frag a0, a1, b;
    const int mrow = lane & 15, hl = lane >> 4;
    a0.q[0] = *(const vi4*)&aTi[mrow * 20 + hl * 4];
    a0.q[1] = *(const vi4*)&aTi[mrow * 20 + 8 + hl * 4];
    a1.q[0] = *(const vi4*)&aTi[(16 + mrow) * 20 + hl * 4];
    a1.q[1] = *(const vi4*)&aTi[(16 + mrow) * 20 + 8 + hl * 4];
    const int nrow = wave * 16 + (lane & 15);
    b.q[0] = *(const vi4*)&bTi[nrow * 20 + hl * 4];
    b.q[1] = *(const vi4*)&bTi[nrow * 20 + 8 + hl * 4];
    c0 = __builtin_amdgcn_wmma_f32_16x16x32_bf16(false, a0.v, false, b.v,
                                                 (short)0, c0, false, false);
    c1 = __builtin_amdgcn_wmma_f32_16x16x32_bf16(false, a1.v, false, b.v,
                                                 (short)0, c1, false, false);
    buf ^= 1;
  }

  // Epilogue: lane holds col ncol; vgpr r holds rows (m0 + r + 8*half) and +16
  const int hl   = lane >> 4;
  const int ncol = n0 + wave * 16 + (lane & 15);
  const float bb = bias ? bias[ncol] : 0.0f;
#pragma unroll
  for (int r = 0; r < 8; ++r) {
    int gm = m0 + r + hl * 8;
    if (gm < M)      C[(size_t)gm * N + ncol]        = c0[r] + bb;
    if (gm + 16 < M) C[(size_t)(gm + 16) * N + ncol] = c1[r] + bb;
  }
}

// ---------------------------------------------------------------------------
// z[i*NI + j, l] = relu( sum_k d[i,k] * s[j,k] * t[l,k] )
// GEMM over MP = ND*NI pair-rows, N = NT padded to 112 (7 active waves), K=HID.
// t tiles TDM-prefetched (OOB rows >= NT return zero via tensor_dim1 = NT).
// ---------------------------------------------------------------------------
__global__ __launch_bounds__(256) void wmma_einsum_kernel(
    const float* __restrict__ d, const float* __restrict__ s,
    const float* __restrict__ t, float* __restrict__ z) {
  __shared__ __align__(16) char smem[38912];
  float* bS0 = (float*)smem;                              // 112x32 fp32
  float* bS1 = (float*)(smem + 14336);
  unsigned short* bH = (unsigned short*)(smem + 28672);   // 112 x 40 bf16
  unsigned short* aH = (unsigned short*)(smem + 37632);   // 16 x 40 bf16
  const int* aTi = (const int*)aH;
  const int* bTi = (const int*)bH;

  const int MP  = ND * NI;
  const int tid = threadIdx.x, lane = tid & 31, wave = tid >> 5;
  const int m0  = blockIdx.x * 16;
  const bool w0 = (__builtin_amdgcn_readfirstlane(wave) == 0);

  v8f c = {};

#if USE_TDM
  if (w0) tdm_load_2d(t, 0u, 32u, 112u, 32u, (unsigned)NT, (unsigned)HID);
#endif

  int buf = 0;
  for (int kt = 0; kt < HID; kt += 32) {
#if USE_TDM
    if (w0) {
      if (kt + 32 < HID) {
        tdm_load_2d(t + kt + 32, buf ? 0u : 14336u, 32u, 112u, 32u,
                    (unsigned)NT, (unsigned)HID);
        __builtin_amdgcn_s_wait_tensorcnt(1);
      } else {
        __builtin_amdgcn_s_wait_tensorcnt(0);
      }
    }
#else
    {
      float* bSd = buf ? bS1 : bS0;
      vf4 vz = {0.0f, 0.0f, 0.0f, 0.0f};
#pragma unroll
      for (int e = 0; e < 4; ++e) {
        int vidx = e * 256 + tid;
        if (vidx < 896) {
          int l = vidx >> 3, k4 = (vidx & 7) << 2;
          vf4 v = (l < NT) ? *(const vf4*)&t[(size_t)l * HID + kt + k4] : vz;
          *(vf4*)&bSd[l * 32 + k4] = v;
        }
      }
    }
#endif
    __syncthreads();

    // A tile: a[m][k] = d[i][k] * s[j][k]; 2 k-adjacent elems -> b32 store
    {
      int m = tid >> 4, k2 = (tid & 15) << 1;
      int gm = m0 + m, gk = kt + k2;
      int gmc = gm < MP ? gm : MP - 1;
      int i = gmc / NI, j = gmc - i * NI;
      float v0 = d[i * HID + gk]     * s[j * HID + gk];
      float v1 = d[i * HID + gk + 1] * s[j * HID + gk + 1];
      if (gm >= MP) { v0 = 0.0f; v1 = 0.0f; }
      *(unsigned*)&aH[m * 40 + k2] = pk2bf(v0, v1);
    }
    // Convert t tile fp32(LDS) -> bf16 fragment layout bH[l*40 + k], b64 stores
    {
      const float* bSc = buf ? bS1 : bS0;
#pragma unroll
      for (int e = 0; e < 4; ++e) {
        int vidx = e * 256 + tid;         // need 896 = 112 x 8 float4s
        if (vidx < 896) {
          int l = vidx >> 3, k4 = (vidx & 7) << 2;
          vf4 v = *(const vf4*)&bSc[l * 32 + k4];
          uint2 p;
          p.x = pk2bf(v[0], v[1]);
          p.y = pk2bf(v[2], v[3]);
          *(uint2*)&bH[l * 40 + k4] = p;
        }
      }
    }
    __syncthreads();

    if (wave < 7) {                       // wave-uniform: EXEC all-1s inside
      Frag a, b;
      const int mrow = lane & 15, hl = lane >> 4;
      a.q[0] = *(const vi4*)&aTi[mrow * 20 + hl * 4];
      a.q[1] = *(const vi4*)&aTi[mrow * 20 + 8 + hl * 4];
      const int lrow = wave * 16 + (lane & 15);
      b.q[0] = *(const vi4*)&bTi[lrow * 20 + hl * 4];
      b.q[1] = *(const vi4*)&bTi[lrow * 20 + 8 + hl * 4];
      c = __builtin_amdgcn_wmma_f32_16x16x32_bf16(false, a.v, false, b.v,
                                                  (short)0, c, false, false);
    }
    buf ^= 1;
  }

  if (wave < 7) {
    const int hl = lane >> 4;
    const int l  = wave * 16 + (lane & 15);
#pragma unroll
    for (int r = 0; r < 8; ++r) {
      int gm = m0 + r + hl * 8;
      if (gm < MP && l < NT)
        z[(size_t)gm * NT + l] = fmaxf(c[r], 0.0f);
    }
  }
}

// ---------------------------------------------------------------------------
// GCN support kernels (deterministic: integer atomics only, gather aggregation)
// ---------------------------------------------------------------------------
__global__ void k_zero_deg(unsigned* deg, int n) {
  int i = blockIdx.x * blockDim.x + threadIdx.x;
  if (i < n) deg[i] = 0u;
}

__global__ void k_count(const int* __restrict__ ei, int E, unsigned* deg, int n) {
  int e = blockIdx.x * blockDim.x + threadIdx.x;
  if (e < E) {
    int c = ei[E + e];                    // target of edge e
    if (c >= 0 && c < n) atomicAdd(&deg[c], 1u);
  }
}

__global__ void k_dis(const unsigned* __restrict__ deg, float* dis, int n) {
  int i = blockIdx.x * blockDim.x + threadIdx.x;
  if (i < n) dis[i] = rsqrtf((float)deg[i] + 1.0f);   // +1 self loop
}

// out[c,:] = (opt relu)( sum_{e: col==c} dis[r]*dis[c]*x[r,:] + dis[c]^2*x[c,:] + b )
__global__ __launch_bounds__(256) void k_aggregate(
    const float* __restrict__ x, const int* __restrict__ ei, int E,
    const float* __restrict__ dis, const float* __restrict__ b,
    float* __restrict__ out, int do_relu) {
  const int c = blockIdx.x;
  const int f = threadIdx.x;              // feature and feature+256
  const float dc = dis[c];
  float a0 = dc * dc * x[(size_t)c * HID + f];
  float a1 = dc * dc * x[(size_t)c * HID + f + 256];
  for (int e = 0; e < E; ++e) {
    if (ei[E + e] == c) {
      int r = ei[e];
      float nm = dis[r] * dc;
      a0 += nm * x[(size_t)r * HID + f];
      a1 += nm * x[(size_t)r * HID + f + 256];
    }
  }
  a0 += b[f];
  a1 += b[f + 256];
  if (do_relu) { a0 = fmaxf(a0, 0.0f); a1 = fmaxf(a1, 0.0f); }
  out[(size_t)c * HID + f]       = a0;
  out[(size_t)c * HID + f + 256] = a1;
}

// ---------------------------------------------------------------------------
// Host orchestration
// ---------------------------------------------------------------------------
static inline int cdiv(int a, int b) { return (a + b - 1) / b; }

static void run_branch(const float* x, const int* ei, int E, int n, int K,
                       const float* Wp, const float* bp,
                       const float* W1, const float* b1,
                       const float* W2, const float* b2,
                       float* h0, float* tmp, float* h1, float* h2,
                       unsigned* deg, float* dis, hipStream_t stream) {
  dim3 blk(256);
  dim3 gP(cdiv(n, 32), HID / 128);
  wmma_gemm_kernel<<<gP, blk, 0, stream>>>(x, Wp, bp, h0, n, K, HID);
  k_zero_deg<<<cdiv(n, 256), blk, 0, stream>>>(deg, n);
  k_count<<<cdiv(E, 256), blk, 0, stream>>>(ei, E, deg, n);
  k_dis<<<cdiv(n, 256), blk, 0, stream>>>(deg, dis, n);
  wmma_gemm_kernel<<<gP, blk, 0, stream>>>(h0, W1, nullptr, tmp, n, HID, HID);
  k_aggregate<<<dim3(n), blk, 0, stream>>>(tmp, ei, E, dis, b1, h1, 1);
  wmma_gemm_kernel<<<gP, blk, 0, stream>>>(h1, W2, nullptr, tmp, n, HID, HID);
  k_aggregate<<<dim3(n), blk, 0, stream>>>(tmp, ei, E, dis, b2, h2, 0);
}

extern "C" void kernel_launch(void* const* d_in, const int* in_sizes, int n_in,
                              void* d_out, int out_size, void* d_ws, size_t ws_size,
                              hipStream_t stream) {
  const int*   g_drug = (const int*)d_in[0];
  const float* x_drug = (const float*)d_in[1];
  const int*   g_dis  = (const int*)d_in[2];
  const float* x_dis  = (const float*)d_in[3];
  const int*   g_tgt  = (const int*)d_in[4];
  const float* x_tgt  = (const float*)d_in[5];

  const float* Wp_d = (const float*)d_in[6];
  const float* bp_d = (const float*)d_in[7];
  const float* W1_d = (const float*)d_in[8];
  const float* b1_d = (const float*)d_in[9];
  const float* W2_d = (const float*)d_in[10];
  const float* b2_d = (const float*)d_in[11];

  const float* Wp_s = (const float*)d_in[12];
  const float* bp_s = (const float*)d_in[13];
  const float* W1_s = (const float*)d_in[14];
  const float* b1_s = (const float*)d_in[15];
  const float* W2_s = (const float*)d_in[16];
  const float* b2_s = (const float*)d_in[17];

  const float* Wp_t = (const float*)d_in[18];
  const float* bp_t = (const float*)d_in[19];
  const float* W1_t = (const float*)d_in[20];
  const float* b1_t = (const float*)d_in[21];
  const float* W2_t = (const float*)d_in[22];
  const float* b2_t = (const float*)d_in[23];

  const int E_drug = in_sizes[0] / 2;
  const int E_dis  = in_sizes[2] / 2;
  const int E_tgt  = in_sizes[4] / 2;
  const int K_drug = in_sizes[1] / ND;   // NI*NT
  const int K_dis  = in_sizes[3] / NI;   // ND*NT
  const int K_tgt  = in_sizes[5] / NT;   // NI*ND

  char* w = (char*)d_ws;
  auto takeF = [&](size_t nf) {
    float* p = (float*)w;
    w += ((nf * sizeof(float) + 15) & ~(size_t)15);
    return p;
  };
  auto takeU = [&](size_t nu) {
    unsigned* p = (unsigned*)w;
    w += ((nu * sizeof(unsigned) + 15) & ~(size_t)15);
    return p;
  };

  float *h0_d = takeF((size_t)ND * HID), *tp_d = takeF((size_t)ND * HID);
  float *h1_d = takeF((size_t)ND * HID), *h2_d = takeF((size_t)ND * HID);
  float *h0_s = takeF((size_t)NI * HID), *tp_s = takeF((size_t)NI * HID);
  float *h1_s = takeF((size_t)NI * HID), *h2_s = takeF((size_t)NI * HID);
  float *h0_t = takeF((size_t)NT * HID), *tp_t = takeF((size_t)NT * HID);
  float *h1_t = takeF((size_t)NT * HID), *h2_t = takeF((size_t)NT * HID);
  unsigned* deg = takeU(NI);   // reused per branch (max n = NI)
  float*    dis = takeF(NI);

  run_branch(x_drug, g_drug, E_drug, ND, K_drug, Wp_d, bp_d, W1_d, b1_d, W2_d, b2_d,
             h0_d, tp_d, h1_d, h2_d, deg, dis, stream);
  run_branch(x_dis, g_dis, E_dis, NI, K_dis, Wp_s, bp_s, W1_s, b1_s, W2_s, b2_s,
             h0_s, tp_s, h1_s, h2_s, deg, dis, stream);
  run_branch(x_tgt, g_tgt, E_tgt, NT, K_tgt, Wp_t, bp_t, W1_t, b1_t, W2_t, b2_t,
             h0_t, tp_t, h1_t, h2_t, deg, dis, stream);

  const int MP = ND * NI;
  wmma_einsum_kernel<<<dim3(cdiv(MP, 16)), dim3(256), 0, stream>>>(
      h2_d, h2_s, h2_t, (float*)d_out);
}
